// CppMegaM2RNNMixer_76012331204912
// MI455X (gfx1250) — matrix operand order repaired and checked by reading, not compiled, not run on hardware
//
#include <hip/hip_runtime.h>

// ---------------- problem dims ----------------
constexpr int SEQ    = 1024;
constexpr int BATCH  = 2;
constexpr int DMODEL = 1024;
constexpr int NH     = 8;
constexpr int KD     = 64;
constexpr int VD     = 64;
constexpr int KER    = 4;
constexpr int CONVD  = KD + KD + NH * VD;      // 640
constexpr int PROJD  = CONVD + NH + NH * VD;   // 1160
constexpr int GOFF   = CONVD + NH;             // 648 (gate offset)
constexpr int GD     = NH * VD;                // 512
constexpr int ROWS   = BATCH * SEQ;            // 2048

// ---------------- WMMA plumbing ----------------
typedef __bf16 v16bf __attribute__((ext_vector_type(16)));
typedef float  v8f   __attribute__((ext_vector_type(8)));
typedef unsigned short u16x16 __attribute__((ext_vector_type(16)));

union Frag {
  u16x16 v;
  unsigned short h[16];
  unsigned int   u[8];
};

__device__ __forceinline__ unsigned short f2bf(float f) {
  unsigned u = __float_as_uint(f);
  u += 0x7FFFu + ((u >> 16) & 1u);        // round-to-nearest-even
  return (unsigned short)(u >> 16);
}

// packed f32x2 -> bf16x2 (one v_cvt_pk_bf16_f32 where available)
__device__ __forceinline__ unsigned pack_bf16x2(float lo, float hi) {
#if __has_builtin(__builtin_amdgcn_cvt_pk_bf16_f32)
  typedef __bf16 v2bf __attribute__((ext_vector_type(2)));
  v2bf r = __builtin_amdgcn_cvt_pk_bf16_f32(lo, hi);
  return __builtin_bit_cast(unsigned, r);
#else
  return (unsigned)f2bf(lo) | ((unsigned)f2bf(hi) << 16);
#endif
}

__device__ __forceinline__ v8f wmma_bf16(const Frag& a, const Frag& b, v8f c) {
  return __builtin_amdgcn_wmma_f32_16x16x32_bf16(
      false, __builtin_bit_cast(v16bf, a.v),
      false, __builtin_bit_cast(v16bf, b.v),
      (short)0, c, false, false);
}

// branch-free tanh: 1 - 2/(exp(2x)+1); exact limits at +/-inf
__device__ __forceinline__ float tanh_fast(float x) {
  float e = __expf(2.f * x);
  return __builtin_fmaf(-2.f, __builtin_amdgcn_rcpf(e + 1.f), 1.f);
}

__device__ __forceinline__ float silu_f(float x) {
  return x * __builtin_amdgcn_rcpf(1.f + __expf(-x));
}

// ---------------- WMMA GEMM: C[MxN] = A[MxK](f32->bf16) * B[KxN](f32->bf16) ----------------
// REMAP_A: logical row r=(b*S+s) is read from physical row (s*Bb+b) of A.
// REMAP_C: logical row r=(b*S+s) is written to physical row (s*Bb+b) of C.
constexpr int BM = 128, BN = 128, BK = 64;
constexpr int LDA_H = BK + 8;   // padded halves per A row
constexpr int LDB_H = BK + 8;   // padded halves per B^T row (K-contiguous)

template <int REMAP_A, int REMAP_C>
__global__ __launch_bounds__(256) void gemm_wmma_kernel(
    const float* __restrict__ A, const float* __restrict__ Bm, float* __restrict__ C,
    int M, int N, int K, int Sdim, int Bb, int ldc)
{
  __shared__ unsigned short sA[BM * LDA_H];
  __shared__ unsigned short sBt[BN * LDB_H];   // transposed: [n][k]

  const int tid  = threadIdx.x;
  const int lane = tid & 31;
  const int w    = tid >> 5;           // 8 waves
  const int wm   = w >> 2;             // 0..1 : 64-row strip
  const int wn   = w & 3;              // 0..3 : 32-col strip
  const int lr   = lane & 15, lh = lane >> 4;
  const int bm   = blockIdx.y * BM;
  const int bn   = blockIdx.x * BN;

  v8f acc[4][2];
#pragma unroll
  for (int mi = 0; mi < 4; ++mi)
#pragma unroll
    for (int nj = 0; nj < 2; ++nj)
      acc[mi][nj] = (v8f){0, 0, 0, 0, 0, 0, 0, 0};

  for (int k0 = 0; k0 < K; k0 += BK) {
    __syncthreads();
    // ---- A tile 128x64 : 2 threads per row, 32 floats each, packed bf16x2 stores ----
    {
      int row  = tid >> 1;
      int c0   = (tid & 1) * 32;
      int lrow = bm + row;
      int prow = REMAP_A ? ((lrow % Sdim) * Bb + (lrow / Sdim)) : lrow;
      const float4* s4 = (const float4*)(A + (size_t)prow * K + k0 + c0);
#pragma unroll
      for (int i = 0; i < 8; ++i) {
        float4 vv = s4[i];
        unsigned* d = (unsigned*)&sA[row * LDA_H + c0 + 4 * i];
        d[0] = pack_bf16x2(vv.x, vv.y);
        d[1] = pack_bf16x2(vv.z, vv.w);
      }
    }
    // ---- B tile 64x128 -> transposed LDS [n][k]: each thread packs 2 K-rows x 16 cols ----
    {
      int kp = tid >> 3;              // 0..31 : K pair index
      int n0 = (tid & 7) * 16;        // 0..112
      const float* r0 = Bm + (size_t)(k0 + 2 * kp) * N + bn + n0;
      const float* r1 = r0 + N;
      float a0[16], a1[16];
      if (bn + n0 + 16 <= N) {
        const float4* p0 = (const float4*)r0;
        const float4* p1 = (const float4*)r1;
#pragma unroll
        for (int i = 0; i < 4; ++i) {
          float4 x = p0[i], y = p1[i];
          a0[4*i+0] = x.x; a0[4*i+1] = x.y; a0[4*i+2] = x.z; a0[4*i+3] = x.w;
          a1[4*i+0] = y.x; a1[4*i+1] = y.y; a1[4*i+2] = y.z; a1[4*i+3] = y.w;
        }
      } else {
#pragma unroll
        for (int i = 0; i < 16; ++i) {
          int nc = bn + n0 + i;
          a0[i] = (nc < N) ? r0[i] : 0.f;
          a1[i] = (nc < N) ? r1[i] : 0.f;
        }
      }
#pragma unroll
      for (int i = 0; i < 16; ++i)
        *(unsigned*)&sBt[(n0 + i) * LDB_H + 2 * kp] = pack_bf16x2(a0[i], a1[i]);
    }
    __syncthreads();
    // ---- two K-chunks of fragments, 16 WMMAs per wave per fill ----
#pragma unroll
    for (int kc = 0; kc < 2; ++kc) {
      Frag af[4], bfv[2];
#pragma unroll
      for (int mi = 0; mi < 4; ++mi) {
        const unsigned short* base = &sA[(wm * 64 + mi * 16 + lr) * LDA_H];
#pragma unroll
        for (int p = 0; p < 8; ++p) {
          int kkp = kc * 32 + ((p < 4) ? 2 * p : 2 * p + 8) + 8 * lh;  // ISA A-layout K map
          af[mi].u[p] = *(const unsigned int*)(base + kkp);
        }
      }
#pragma unroll
      for (int nj = 0; nj < 2; ++nj) {
        const unsigned short* base = &sBt[(wn * 32 + nj * 16 + lr) * LDB_H];
#pragma unroll
        for (int p = 0; p < 8; ++p) {
          int kkp = kc * 32 + ((p < 4) ? 2 * p : 2 * p + 8) + 8 * lh;  // B: same ISA K map as A
          bfv[nj].u[p] = *(const unsigned int*)(base + kkp);
        }
      }
#pragma unroll
      for (int mi = 0; mi < 4; ++mi)
#pragma unroll
        for (int nj = 0; nj < 2; ++nj)
          acc[mi][nj] = wmma_bf16(af[mi], bfv[nj], acc[mi][nj]);
    }
  }

  // ---- epilogue (C layout: lane=col, vgpr r -> M = r + 8*lh) ----
#pragma unroll
  for (int mi = 0; mi < 4; ++mi)
#pragma unroll
    for (int nj = 0; nj < 2; ++nj) {
      int col = bn + wn * 32 + nj * 16 + lr;
      if (col < N) {
#pragma unroll
        for (int r = 0; r < 8; ++r) {
          int lrow = bm + wm * 64 + mi * 16 + r + 8 * lh;
          int prow = REMAP_C ? ((lrow % Sdim) * Bb + (lrow / Sdim)) : lrow;
          C[(size_t)prow * ldc + col] = acc[mi][nj][r];
        }
      }
    }
}

// ---------------- depthwise causal conv (K=4) + SiLU ----------------
__global__ void conv_silu_kernel(const float* __restrict__ proj,
                                 const float* __restrict__ conv_w,
                                 const float* __restrict__ conv_b,
                                 float* __restrict__ convAct)
{
  int idx = blockIdx.x * blockDim.x + threadIdx.x;
  if (idx >= ROWS * CONVD) return;
  int c  = idx % CONVD;
  int bs = idx / CONVD;
  int s  = bs % SEQ;
  int b  = bs / SEQ;
  float acc = conv_b[c];
#pragma unroll
  for (int j = 0; j < KER; ++j) {
    int tt = s - (KER - 1) + j;
    if (tt >= 0)
      acc += conv_w[c * KER + j] * proj[((size_t)b * SEQ + tt) * PROJD + c];
  }
  convAct[idx] = silu_f(acc);
}

// ---------------- per-head forget gate f = exp(-exp(A_log)*softplus(f_in+dt_bias)) ----------------
__global__ void f_gate_kernel(const float* __restrict__ proj,
                              const float* __restrict__ A_log,
                              const float* __restrict__ dt_bias,
                              float* __restrict__ fbuf)
{
  int idx = blockIdx.x * blockDim.x + threadIdx.x;
  if (idx >= ROWS * NH) return;
  int n  = idx % NH;
  int bs = idx / NH;
  float x  = proj[(size_t)bs * PROJD + CONVD + n] + dt_bias[n];
  float sp = (x > 20.f) ? x : log1pf(expf(x));
  fbuf[idx] = expf(-expf(A_log[n]) * sp);
}

// ---------------- sequential tanh-RNN scan: 1 block per (batch,head) ----------------
// h (64x64) lives in C-layout f32 registers (wave-private 16-row strips);
// an A-layout bf16 shadow copy in LDS feeds next step's WMMA A fragments.
// state_W lives in per-wave B fragments (registers), loaded once.
constexpr int LDH_H = 72;   // padded halves per h row in bf16 shadow

__global__ __launch_bounds__(128) void scan_kernel(
    const float* __restrict__ convAct, const float* __restrict__ fbuf,
    const float* __restrict__ state_W, const float* __restrict__ Dmat,
    float* __restrict__ preGate)
{
  const int bnid = blockIdx.x;
  const int b = bnid / NH, n = bnid % NH;

  __shared__ unsigned short ldsHb[64 * LDH_H];  // bf16 h, A layout [m][k]
  __shared__ float qs[64], ks[64], vs[64];
  __shared__ float ySum[64];
  __shared__ float fsh;

  const int tid  = threadIdx.x;
  const int lane = tid & 31;
  const int w    = tid >> 5;        // 4 waves
  const int lr   = lane & 15, lh = lane >> 4;
  const int mbase = w * 16;

  for (int i = tid; i < 64 * LDH_H; i += 128) ldsHb[i] = 0;

  // h state in registers, C layout: hreg[j][r] = h[mbase + r + 8*lh][j*16 + lr]
  float hreg[4][8];
#pragma unroll
  for (int j = 0; j < 4; ++j)
#pragma unroll
    for (int r = 0; r < 8; ++r) hreg[j][r] = 0.f;

  // preload state_W[n] (K=u rows, N=v cols) into 8 B fragments, once.
  Frag bfr[2][4];
  const float* sw = state_W + (size_t)n * KD * VD;
#pragma unroll
  for (int kc = 0; kc < 2; ++kc)
#pragma unroll
    for (int j = 0; j < 4; ++j) {
      int col = j * 16 + lr;
#pragma unroll
      for (int p = 0; p < 8; ++p) {
        int k = kc * 32 + ((p < 4) ? 2 * p : 2 * p + 8) + 8 * lh;
        bfr[kc][j].u[p] = pack_bf16x2(sw[k * VD + col], sw[(k + 1) * VD + col]);
      }
    }
  float Dv = (tid >= 64) ? Dmat[n * VD + (tid - 64)] : 0.f;
  __syncthreads();

  for (int t = 0; t < SEQ; ++t) {
    const size_t base = (size_t)b * SEQ + t;
    float vcontrib = 0.f;
    if (tid < 64) {
      qs[tid]   = convAct[base * CONVD + tid];
      ks[tid]   = convAct[base * CONVD + KD + tid];
      ySum[tid] = 0.f;   // safe: only this thread read ySum[tid] last step
    } else {
      int i = tid - 64;
      float vv = convAct[base * CONVD + 2 * KD + n * VD + i];
      vs[i] = vv;
      vcontrib = vv * Dv;                       // v*D folded into ySum
    }
    if (tid == 0) {
      fsh = fbuf[base * NH + n];
      if (t + 1 < SEQ) {                        // prefetch next step's q/k/v
        const float* nxt = convAct + (base + 1) * CONVD;
        __builtin_prefetch(nxt,       0, 0);
        __builtin_prefetch(nxt + 32,  0, 0);
        __builtin_prefetch(nxt + 64,  0, 0);
        __builtin_prefetch(nxt + 96,  0, 0);
        __builtin_prefetch(nxt + 2 * KD + n * VD,      0, 0);
        __builtin_prefetch(nxt + 2 * KD + n * VD + 32, 0, 0);
      }
    }
    __syncthreads();                            // barrier #1
    const float f = fsh, omf = 1.f - fsh;

    // A fragments: pure packed bf16 loads from the shadow copy (no conversion)
    Frag afr[2];
    {
      const unsigned short* hrow = &ldsHb[(mbase + lr) * LDH_H];
#pragma unroll
      for (int kc = 0; kc < 2; ++kc)
#pragma unroll
        for (int p = 0; p < 8; ++p) {
          int kb = kc * 32 + ((p < 4) ? 2 * p : 2 * p + 8) + 8 * lh;
          afr[kc].u[p] = *(const unsigned int*)(hrow + kb);
        }
    }

    // hoist this lane's k/q row values (shared across the 4 column tiles)
    float kk[8], qq[8];
#pragma unroll
    for (int r = 0; r < 8; ++r) {
      int mr = mbase + r + 8 * lh;
      kk[r] = ks[mr];
      qq[r] = qs[mr];
    }

#pragma unroll
    for (int j = 0; j < 4; ++j) {
      const int col = j * 16 + lr;
      const float vv = vs[col];
      v8f acc;
#pragma unroll
      for (int r = 0; r < 8; ++r)               // C init = k ⊗ v (rank-1 input)
        acc[r] = kk[r] * vv;
      acc = wmma_bf16(afr[0], bfr[0][j], acc);  // h @ state_W (K=0..31)
      acc = wmma_bf16(afr[1], bfr[1][j], acc);  // h @ state_W (K=32..63)
      float part = 0.f;
#pragma unroll
      for (int r = 0; r < 8; ++r) {
        float hv = f * hreg[j][r] + omf * tanh_fast(acc[r]);
        hreg[j][r] = hv;                        // state stays in registers
        ldsHb[(mbase + r + 8 * lh) * LDH_H + col] = f2bf(hv);  // bf16 shadow (wave-private rows)
        part += qq[r] * hv;                     // y[col] += q[m]*h[m][col]
      }
      atomicAdd(&ySum[col], part);              // ds_add_f32 cross-wave reduce
    }
    if (tid >= 64) atomicAdd(&ySum[tid - 64], vcontrib);
    __syncthreads();                            // barrier #2
    if (tid < 64)
      preGate[base * GD + n * VD + tid] = ySum[tid];
  }
}

// ---------------- SiLU(g) gating + RMS norm ----------------
__global__ __launch_bounds__(256) void gate_rms_kernel(
    const float* __restrict__ preGate, const float* __restrict__ proj,
    const float* __restrict__ norm_w, float* __restrict__ normed)
{
  const int row = blockIdx.x;                   // b*SEQ + s
  __shared__ float red[256];
  const int tid = threadIdx.x;
  float o[2];
#pragma unroll
  for (int u = 0; u < 2; ++u) {
    int i = tid + u * 256;
    float y = preGate[(size_t)row * GD + i];
    float g = proj[(size_t)row * PROJD + GOFF + i];
    o[u] = y * silu_f(g);
  }
  red[tid] = o[0] * o[0] + o[1] * o[1];
  __syncthreads();
  for (int st = 128; st > 0; st >>= 1) {
    if (tid < st) red[tid] += red[tid + st];
    __syncthreads();
  }
  float scale = rsqrtf(red[0] / (float)GD + 1e-5f);
#pragma unroll
  for (int u = 0; u < 2; ++u) {
    int i = tid + u * 256;
    normed[(size_t)row * GD + i] = o[u] * scale * norm_w[i];
  }
}

// ---------------- launcher ----------------
extern "C" void kernel_launch(void* const* d_in, const int* in_sizes, int n_in,
                              void* d_out, int out_size, void* d_ws, size_t ws_size,
                              hipStream_t stream) {
  const float* hidden  = (const float*)d_in[0];
  const float* W_in    = (const float*)d_in[1];
  const float* A_log   = (const float*)d_in[2];
  const float* dt_bias = (const float*)d_in[3];
  const float* conv_w  = (const float*)d_in[4];
  const float* conv_b  = (const float*)d_in[5];
  const float* state_W = (const float*)d_in[6];
  const float* Dmat    = (const float*)d_in[7];
  const float* norm_w  = (const float*)d_in[8];
  const float* W_out   = (const float*)d_in[9];
  float* out = (float*)d_out;

  float* ws      = (float*)d_ws;
  float* proj    = ws;                                   // ROWS*PROJD
  float* convAct = proj    + (size_t)ROWS * PROJD;       // ROWS*CONVD
  float* fbuf    = convAct + (size_t)ROWS * CONVD;       // ROWS*NH
  float* preGate = fbuf    + (size_t)ROWS * NH;          // ROWS*GD
  float* normed  = preGate + (size_t)ROWS * GD;          // ROWS*GD

  // 1) proj = hidden(s,b,:) @ W_in   (rows remapped (b,s)->phys(s,b))
  dim3 g1((PROJD + BN - 1) / BN, ROWS / BM);
  gemm_wmma_kernel<1, 0><<<g1, 256, 0, stream>>>(
      hidden, W_in, proj, ROWS, PROJD, DMODEL, SEQ, BATCH, PROJD);

  // 2) causal depthwise conv + SiLU
  int nconv = ROWS * CONVD;
  conv_silu_kernel<<<(nconv + 255) / 256, 256, 0, stream>>>(proj, conv_w, conv_b, convAct);

  // 3) forget gate
  int nf = ROWS * NH;
  f_gate_kernel<<<(nf + 255) / 256, 256, 0, stream>>>(proj, A_log, dt_bias, fbuf);

  // 4) sequential WMMA scan: 16 blocks = (batch,head) pairs
  scan_kernel<<<BATCH * NH, 128, 0, stream>>>(convAct, fbuf, state_W, Dmat, preGate);

  // 5) SiLU(g) gating + RMS norm
  gate_rms_kernel<<<ROWS, 256, 0, stream>>>(preGate, proj, norm_w, normed);

  // 6) out(s,b,:) = normed @ W_out   (stores remapped (b,s)->phys(s,b))
  dim3 g2(DMODEL / BN, ROWS / BM);
  gemm_wmma_kernel<0, 1><<<g2, 256, 0, stream>>>(
      normed, W_out, out, ROWS, DMODEL, GD, SEQ, BATCH, DMODEL);
}
